// mamba_model_82557861364405
// MI455X (gfx1250) — compile-verified, matrix operand order checked by reading, not run on hardware
//
#include <hip/hip_runtime.h>
#include <hip/hip_bf16.h>

// ---------------------------------------------------------------------------
// Mamba (2 layers) for MI455X / gfx1250, wave32, WMMA bf16 tensor path.
// ---------------------------------------------------------------------------

#define DEPTH 2
#define BATCH 4
#define LSEQ  4096
#define DM    512
#define DI    1024
#define DS    16
#define DTR   32
#define KW    4
#define MROWS (BATCH * LSEQ)   // 16384

typedef unsigned short u16;
typedef unsigned int   u32;

typedef __attribute__((ext_vector_type(16))) __bf16 v16bf;
typedef __attribute__((ext_vector_type(8)))  float  v8f;

union FragU {
    uint4 q[2];
    v16bf v;
};

__device__ __forceinline__ u16 f2bf(float f) {
    u32 u = __float_as_uint(f);
    u32 r = u + 0x7FFFu + ((u >> 16) & 1u);   // round-to-nearest-even
    return (u16)(r >> 16);
}

// numerically stable fast softplus: max(x,0) + log(1 + exp(-|x|))
__device__ __forceinline__ float fast_softplus(float x) {
    return fmaxf(x, 0.0f) + __logf(1.0f + __expf(-fabsf(x)));
}

// ---------------------------------------------------------------------------
// fp32 -> bf16 elementwise convert (weights, small tensors)
// ---------------------------------------------------------------------------
__global__ __launch_bounds__(256) void cvt_bf16_kernel(const float* __restrict__ src,
                                                       u16* __restrict__ dst, int n) {
    int i = blockIdx.x * 256 + threadIdx.x;
    if (i < n) dst[i] = f2bf(src[i]);
}

// ---------------------------------------------------------------------------
// LayerNorm over DM=512, one block per row, bf16 output for WMMA A operand
// ---------------------------------------------------------------------------
__global__ __launch_bounds__(256) void layernorm_bf16_kernel(const float* __restrict__ h,
                                                             const float* __restrict__ w,
                                                             const float* __restrict__ b,
                                                             u16* __restrict__ out) {
    __shared__ float sh_s[256];
    __shared__ float sh_q[256];
    const int row = blockIdx.x;
    const int tid = threadIdx.x;
    const float* hp = h + (size_t)row * DM;
    float v0 = hp[tid];
    float v1 = hp[tid + 256];
    sh_s[tid] = v0 + v1;
    sh_q[tid] = v0 * v0 + v1 * v1;
    __syncthreads();
    for (int off = 128; off > 0; off >>= 1) {
        if (tid < off) {
            sh_s[tid] += sh_s[tid + off];
            sh_q[tid] += sh_q[tid + off];
        }
        __syncthreads();
    }
    const float mean = sh_s[0] * (1.0f / DM);
    const float var  = sh_q[0] * (1.0f / DM) - mean * mean;
    const float inv  = rsqrtf(var + 1e-5f);
    out[(size_t)row * DM + tid]       = f2bf((v0 - mean) * inv * w[tid] + b[tid]);
    out[(size_t)row * DM + tid + 256] = f2bf((v1 - mean) * inv * w[tid + 256] + b[tid + 256]);
}

// ---------------------------------------------------------------------------
// WMMA bf16 GEMM:  C[m,n] = sum_k A[m,k] * W[n,k]   (+ optional softplus bias)
//   A: (M x K) bf16 row-major, W: (N x K) bf16 row-major, C: (M x N) fp32.
//   Requires M%256==0, N%64==0, K%32==0. Block = 128 thr = 4 waves stacked in M,
//   each wave computes a 64x64 tile = 4x4 v_wmma_f32_16x16x32_bf16 tiles
//   (16 WMMA : 16 B128 loads per K-step).
//   mode 0: plain store;  mode 1: C = softplus(C + bias[n]).
// ---------------------------------------------------------------------------
__global__ __launch_bounds__(128) void wmma_gemm_kernel(const u16* __restrict__ A,
                                                        const u16* __restrict__ W,
                                                        float* __restrict__ C,
                                                        int K, int N,
                                                        const float* __restrict__ bias,
                                                        int mode) {
    const int wave = threadIdx.x >> 5;
    const int lane = threadIdx.x & 31;
    const int m_base = blockIdx.x * 256 + wave * 64;
    const int n_base = blockIdx.y * 64;
    const int r16   = lane & 15;   // row within 16 (M or N index inside tile)
    const int khalf = lane >> 4;   // selects K sub-group per ISA A/B layout

    v8f acc[4][4] = {};

    for (int k0 = 0; k0 < K; k0 += 32) {
        // A fragments: per-lane K window is [k0 + khalf*8, +8) and [k0+16+khalf*8, +8)
        v16bf afrag[4];
#pragma unroll
        for (int i = 0; i < 4; ++i) {
            const u16* ap = A + (size_t)(m_base + i * 16 + r16) * K + k0 + khalf * 8;
            FragU f;
            f.q[0] = *(const uint4*)(ap);        // VGPR 0..3  (K = base .. base+7)
            f.q[1] = *(const uint4*)(ap + 16);   // VGPR 4..7  (K = base+16 .. +23)
            afrag[i] = f.v;
        }
        // B fragments (mirrored layout, N takes the role of M; W row-major over K)
        v16bf bfrag[4];
#pragma unroll
        for (int j = 0; j < 4; ++j) {
            const u16* wp = W + (size_t)(n_base + j * 16 + r16) * K + k0 + khalf * 8;
            FragU f;
            f.q[0] = *(const uint4*)(wp);
            f.q[1] = *(const uint4*)(wp + 16);
            bfrag[j] = f.v;
        }
#pragma unroll
        for (int i = 0; i < 4; ++i)
#pragma unroll
            for (int j = 0; j < 4; ++j)
                acc[i][j] = __builtin_amdgcn_wmma_f32_16x16x32_bf16(
                    false, afrag[i], false, bfrag[j], (short)0, acc[i][j], false, false);
    }

    // Epilogue. C-layout: element v of the v8f lives at m = v + 8*(lane/16), n = lane%16.
#pragma unroll
    for (int i = 0; i < 4; ++i) {
#pragma unroll
        for (int j = 0; j < 4; ++j) {
            const int n = n_base + j * 16 + r16;
#pragma unroll
            for (int v = 0; v < 8; ++v) {
                const int m = m_base + i * 16 + khalf * 8 + v;
                float val = acc[i][j][v];
                if (mode == 1) val = fast_softplus(val + bias[n]);
                C[(size_t)m * N + n] = val;
            }
        }
    }
}

// ---------------------------------------------------------------------------
// Depthwise causal conv (K=4) + bias + SiLU.  xz is (M, 2*DI); xq = cols [0,DI).
// Writes fp32 (for the scan) and bf16 (for the x_proj WMMA GEMM).
// ---------------------------------------------------------------------------
__global__ __launch_bounds__(256) void conv_silu_kernel(const float* __restrict__ xz,
                                                        const float* __restrict__ cw,
                                                        const float* __restrict__ cb,
                                                        float* __restrict__ xqf,
                                                        u16* __restrict__ xqb) {
    const int idx = blockIdx.x * 256 + threadIdx.x;   // over MROWS * DI
    const int c = idx & (DI - 1);
    const int m = idx >> 10;                          // DI = 1024
    const int l = m & (LSEQ - 1);

    float acc = 0.0f;
#pragma unroll
    for (int j = 0; j < KW; ++j) {
        const int ls = l - (KW - 1) + j;
        if (ls >= 0)
            acc += cw[c * KW + j] * xz[(size_t)(m - (KW - 1) + j) * (2 * DI) + c];
    }
    acc += cb[c];
    const float s = acc / (1.0f + __expf(-acc));       // SiLU
    xqf[(size_t)m * DI + c] = s;
    xqb[(size_t)m * DI + c] = f2bf(s);
}

// ---------------------------------------------------------------------------
// Extract dtr (first DTR=32 columns of dbl (M,64)) as bf16 for the dt GEMM.
// ---------------------------------------------------------------------------
__global__ __launch_bounds__(256) void extract_dtr_kernel(const float* __restrict__ dbl,
                                                          u16* __restrict__ dtr) {
    const int idx = blockIdx.x * 256 + threadIdx.x;   // over MROWS * DTR
    const int r = idx & (DTR - 1);
    const int m = idx >> 5;
    dtr[idx] = f2bf(dbl[(size_t)m * 64 + r]);
}

// ---------------------------------------------------------------------------
// Selective scan: one thread per (b, d) channel; 16-wide state in registers.
// Fuses  y = (scan_y + x*D) * silu(z)  and emits bf16 for the out-proj GEMM.
// Bc/Cc loads are wave-uniform (all lanes share (b,l)) -> broadcast from cache.
// ---------------------------------------------------------------------------
__global__ __launch_bounds__(256) void scan_kernel(const float* __restrict__ dt,
                                                   const float* __restrict__ xq,
                                                   const float* __restrict__ dbl,
                                                   const float* __restrict__ xz,
                                                   const float* __restrict__ A_log,
                                                   const float* __restrict__ Dp,
                                                   u16* __restrict__ yact) {
    const int t = blockIdx.x * 256 + threadIdx.x;     // 0 .. BATCH*DI
    const int d = t & (DI - 1);
    const int b = t >> 10;

    float a[DS], h[DS];
#pragma unroll
    for (int s = 0; s < DS; ++s) {
        a[s] = -__expf(A_log[d * DS + s]);
        h[s] = 0.0f;
    }
    const float Dd = Dp[d];

    for (int l = 0; l < LSEQ; ++l) {
        const size_t m = (size_t)b * LSEQ + l;
        const float dtv = dt[m * DI + d];
        const float xv  = xq[m * DI + d];
        const float* bc = dbl + m * 64 + DTR;     // B_c[0..15]
        const float* cc = bc + DS;                // C_c[0..15]
        if (l + 1 < LSEQ)
            __builtin_prefetch(dbl + (m + 1) * 64 + DTR, 0, 1);  // global_prefetch_b8

        const float dx = dtv * xv;
        float y = 0.0f;
#pragma unroll
        for (int s = 0; s < DS; ++s) {
            h[s] = h[s] * __expf(dtv * a[s]) + dx * bc[s];
            y += h[s] * cc[s];
        }
        const float zv  = xz[m * (2 * DI) + DI + d];
        const float sil = zv / (1.0f + __expf(-zv));
        yact[m * DI + d] = f2bf((y + xv * Dd) * sil);
    }
}

// ---------------------------------------------------------------------------
// Host launch
// ---------------------------------------------------------------------------
extern "C" void kernel_launch(void* const* d_in, const int* in_sizes, int n_in,
                              void* d_out, int out_size, void* d_ws, size_t ws_size,
                              hipStream_t stream) {
    const float* x_in   = (const float*)d_in[0];
    const float* ln_w   = (const float*)d_in[1];
    const float* ln_b   = (const float*)d_in[2];
    const float* in_w   = (const float*)d_in[3];
    const float* conv_w = (const float*)d_in[4];
    const float* conv_b = (const float*)d_in[5];
    const float* xproj  = (const float*)d_in[6];
    const float* dt_w   = (const float*)d_in[7];
    const float* dt_b   = (const float*)d_in[8];
    const float* A_log  = (const float*)d_in[9];
    const float* Dp     = (const float*)d_in[10];
    const float* out_w  = (const float*)d_in[11];

    char* ws = (char*)d_ws;
    size_t off = 0;
    auto alloc = [&](size_t bytes) -> void* {
        void* p = ws + off;
        off += (bytes + 255) & ~(size_t)255;
        return p;
    };
    u16*   hn    = (u16*)  alloc((size_t)MROWS * DM * 2);
    float* xz    = (float*)alloc((size_t)MROWS * 2 * DI * 4);
    float* xqf   = (float*)alloc((size_t)MROWS * DI * 4);
    u16*   xqb   = (u16*)  alloc((size_t)MROWS * DI * 2);
    float* dbl   = (float*)alloc((size_t)MROWS * 64 * 4);
    u16*   dtrb  = (u16*)  alloc((size_t)MROWS * DTR * 2);
    float* dtbuf = (float*)alloc((size_t)MROWS * DI * 4);
    u16*   yact  = (u16*)  alloc((size_t)MROWS * DI * 2);
    u16*   winb  = (u16*)  alloc((size_t)2 * DI * DM * 2);
    u16*   wxpb  = (u16*)  alloc((size_t)(DTR + 2 * DS) * DI * 2);
    u16*   wdtb  = (u16*)  alloc((size_t)DI * DTR * 2);
    u16*   woutb = (u16*)  alloc((size_t)DM * DI * 2);

    float* hbuf = (float*)d_out;   // inter-layer activation lives in d_out

    for (int i = 0; i < DEPTH; ++i) {
        // weights -> bf16
        {
            int n;
            n = 2 * DI * DM;
            cvt_bf16_kernel<<<(n + 255) / 256, 256, 0, stream>>>(in_w + (size_t)i * n, winb, n);
            n = (DTR + 2 * DS) * DI;
            cvt_bf16_kernel<<<(n + 255) / 256, 256, 0, stream>>>(xproj + (size_t)i * n, wxpb, n);
            n = DI * DTR;
            cvt_bf16_kernel<<<(n + 255) / 256, 256, 0, stream>>>(dt_w + (size_t)i * n, wdtb, n);
            n = DM * DI;
            cvt_bf16_kernel<<<(n + 255) / 256, 256, 0, stream>>>(out_w + (size_t)i * n, woutb, n);
        }

        const float* hin = (i == 0) ? x_in : (const float*)hbuf;

        // 1) LayerNorm -> bf16
        layernorm_bf16_kernel<<<MROWS, 256, 0, stream>>>(hin, ln_w + i * DM, ln_b + i * DM, hn);

        // 2) in_proj: xz = hn @ in_w^T   (16384 x 2048, K=512)
        wmma_gemm_kernel<<<dim3(MROWS / 256, (2 * DI) / 64), 128, 0, stream>>>(
            hn, winb, xz, DM, 2 * DI, nullptr, 0);

        // 3) depthwise causal conv + SiLU
        conv_silu_kernel<<<(MROWS * DI) / 256, 256, 0, stream>>>(
            xz, conv_w + (size_t)i * DI * KW, conv_b + i * DI, xqf, xqb);

        // 4) x_proj: dbl = xq @ xproj_w^T   (16384 x 64, K=1024)
        wmma_gemm_kernel<<<dim3(MROWS / 256, 64 / 64), 128, 0, stream>>>(
            xqb, wxpb, dbl, DI, DTR + 2 * DS, nullptr, 0);

        // 5) dtr slice -> bf16
        extract_dtr_kernel<<<(MROWS * DTR) / 256, 256, 0, stream>>>(dbl, dtrb);

        // 6) dt = softplus(dtr @ dt_w^T + dt_b)   (16384 x 1024, K=32)
        wmma_gemm_kernel<<<dim3(MROWS / 256, DI / 64), 128, 0, stream>>>(
            dtrb, wdtb, dtbuf, DTR, DI, dt_b + i * DI, 1);

        // 7) selective scan + gating -> yact (bf16)
        scan_kernel<<<(BATCH * DI) / 256, 256, 0, stream>>>(
            dtbuf, xqf, dbl, xz, A_log + (size_t)i * DI * DS, Dp + i * DI, yact);

        // 8) out_proj: h = yact @ out_w^T   (16384 x 512, K=1024)
        wmma_gemm_kernel<<<dim3(MROWS / 256, DM / 64), 128, 0, stream>>>(
            yact, woutb, hbuf, DI, DM, nullptr, 0);
    }
}
